// MultiHeadAttention_76948634075345
// MI455X (gfx1250) — matrix operand order repaired and checked by reading, not compiled, not run on hardware
//
#include <hip/hip_runtime.h>
#include <hip/hip_bf16.h>
#include <stdint.h>

typedef __attribute__((ext_vector_type(8)))  float   v8f;
typedef __attribute__((ext_vector_type(4)))  __bf16  v4bf;
typedef __attribute__((ext_vector_type(8)))  __bf16  v8bf;
typedef __attribute__((ext_vector_type(16))) __bf16  v16bf;

#define BB   4
#define SS   2048
#define DD   1024
#define HH   16
#define DEP  64
#define NHB  64                      // H*B
#define NEGV (-4294967295.0f)        // -2^32+1

static __device__ __forceinline__ __bf16 f2bf(float f) {
  uint32_t u = __builtin_bit_cast(uint32_t, f);
  u += 0x7FFFu + ((u >> 16) & 1u);             // round-to-nearest-even
  uint16_t h = (uint16_t)(u >> 16);
  return __builtin_bit_cast(__bf16, h);
}
static __device__ __forceinline__ float bf2f(__bf16 b) {
  uint16_t h = __builtin_bit_cast(uint16_t, b);
  uint32_t u = ((uint32_t)h) << 16;
  return __builtin_bit_cast(float, u);
}
static __device__ __forceinline__ v16bf cat8(v8bf lo, v8bf hi) {
  return __builtin_shufflevector(lo, hi, 0,1,2,3,4,5,6,7,8,9,10,11,12,13,14,15);
}
#define WMMA_BF16(A_, B_, C_) \
  __builtin_amdgcn_wmma_f32_16x16x32_bf16(false, (A_), false, (B_), (short)0, (C_), false, false)

// ---------------------------------------------------------------------------
// f32 -> bf16 elementwise (n multiple of 4)
__global__ void cvt_x_kernel(const float* __restrict__ x, __bf16* __restrict__ y, int n) {
  int i = (blockIdx.x * blockDim.x + threadIdx.x) * 4;
  if (i >= n) return;
  const float4 f = *(const float4*)(x + i);
  v4bf o;
  o[0] = f2bf(f.x); o[1] = f2bf(f.y); o[2] = f2bf(f.z); o[3] = f2bf(f.w);
  *(v4bf*)(y + i) = o;
}

// W (k-major, DxD) -> Wt bf16 (c-major). LDS-tiled transpose, coalesced both sides.
__global__ __launch_bounds__(256) void cvt_wt_kernel(const float* __restrict__ w,
                                                     __bf16* __restrict__ wt) {
  __shared__ float tile[32][33];
  const int tx = threadIdx.x & 31;
  const int ty = threadIdx.x >> 5;             // 0..7
  const int c0 = blockIdx.x << 5;
  const int k0 = blockIdx.y << 5;
  #pragma unroll
  for (int i = 0; i < 4; ++i)
    tile[ty + 8 * i][tx] = w[(size_t)(k0 + ty + 8 * i) * DD + (c0 + tx)];
  __syncthreads();
  #pragma unroll
  for (int i = 0; i < 4; ++i)
    wt[(size_t)(c0 + ty + 8 * i) * DD + (k0 + tx)] = f2bf(tile[tx][ty + 8 * i]);
}

// ---------------------------------------------------------------------------
// Projection GEMM: Y = X(8192x1024)bf16 @ W(1024x1024) + b.
// Ping-pong register double buffer (2x unrolled K loop): no inter-set copies,
// no WMMA->VALU WAR hazard NOPs, partial s_wait_loadcnt pipelining.
// One wave -> 16x64 tile. vmode=0: store (n,s,64); vmode=1: store V^T (n,64,s).
__global__ __launch_bounds__(256) void proj_gemm_kernel(
    const __bf16* __restrict__ Xbf, const __bf16* __restrict__ Wt,
    const float* __restrict__ bias, __bf16* __restrict__ outp, int vmode) {
  const int wave = threadIdx.x >> 5;
  const int lane = threadIdx.x & 31;
  const int wid  = blockIdx.x * 8 + wave;           // 0..8191 wave tiles
  const int rt   = wid >> 4;                        // 512 row tiles of 16
  const int ct   = wid & 15;                        // 16 col tiles of 64
  const int half = lane >> 4, l16 = lane & 15;
  const int rbase = rt << 4, cbase = ct << 6;

  v8f acc[4] = {};
  const __bf16* xrow  = Xbf + (size_t)(rbase + l16) * DD + half * 16;
  const __bf16* wbase = Wt + (size_t)(cbase + l16) * DD + half * 16;

#define LOAD_FRAGS(kc_, A_, B_) do {                                         \
    const int _k = (kc_) * 32;                                               \
    A_ = cat8(*(const v8bf*)(xrow + _k), *(const v8bf*)(xrow + _k + 8));     \
    B_[0] = *(const v16bf*)(wbase + 0 * 16 * DD + _k);                       \
    B_[1] = *(const v16bf*)(wbase + 1 * 16 * DD + _k);                       \
    B_[2] = *(const v16bf*)(wbase + 2 * 16 * DD + _k);                       \
    B_[3] = *(const v16bf*)(wbase + 3 * 16 * DD + _k);                       \
  } while (0)

  v16bf a0, a1;
  v16bf b0[4], b1[4];
  LOAD_FRAGS(0, a0, b0);
  for (int kc = 0; kc < 32; kc += 2) {
    LOAD_FRAGS(kc + 1, a1, b1);                     // fill set 1
    #pragma unroll
    for (int j = 0; j < 4; ++j)                     // compute on set 0
      acc[j] = WMMA_BF16(a0, b0[j], acc[j]);
    if (kc + 2 < 32)
      LOAD_FRAGS(kc + 2, a0, b0);                   // refill set 0
    #pragma unroll
    for (int j = 0; j < 4; ++j)                     // compute on set 1
      acc[j] = WMMA_BF16(a1, b1[j], acc[j]);
  }
#undef LOAD_FRAGS

  #pragma unroll
  for (int j = 0; j < 4; ++j) {
    const int c = cbase + j * 16 + l16;
    const float bv = bias[c];
    const int hh = c >> 6, d = c & 63;
    #pragma unroll
    for (int r = 0; r < 8; ++r) {
      const int row  = rbase + r + half * 8;        // C layout: M = r + 8*(lane/16)
      const int bidx = row >> 11, s = row & (SS - 1);
      const int nn   = hh * BB + bidx;
      const float val = acc[j][r] + bv;
      const size_t addr = vmode ? (((size_t)nn * DEP + d) * SS + s)
                                : (((size_t)nn * SS + s) * DEP + d);
      outp[addr] = f2bf(val);
    }
  }
}

// ---------------------------------------------------------------------------
// key_mask / query_mask = sign(|sum over depth|)
__global__ void mask_kernel(const __bf16* __restrict__ Qh, const __bf16* __restrict__ Kh,
                            float* __restrict__ qm, float* __restrict__ km) {
  int id = blockIdx.x * blockDim.x + threadIdx.x;
  if (id >= NHB * SS) return;
  const __bf16* q = Qh + (size_t)id * DEP;
  const __bf16* k = Kh + (size_t)id * DEP;
  float sq = 0.f, sk = 0.f;
  #pragma unroll
  for (int d = 0; d < DEP; ++d) { sq += bf2f(q[d]); sk += bf2f(k[d]); }
  qm[id] = (sq != 0.f) ? 1.f : 0.f;
  km[id] = (sk != 0.f) ? 1.f : 0.f;
}

// ---------------------------------------------------------------------------
// Flash attention: one wave per (n, 16-query tile); kv steps of 32 keys.
// K tiles streamed into LDS with double-buffered async copies (ASYNCcnt).
__global__ __launch_bounds__(128) void attn_fa_kernel(
    const __bf16* __restrict__ Qh, const __bf16* __restrict__ Kh,
    const __bf16* __restrict__ VhT, const float* __restrict__ kmask,
    const float* __restrict__ qmask, float* __restrict__ out) {
  __shared__ alignas(16) __bf16 plds[4][16 * 40];       // per-wave P tile (padded)
  __shared__ alignas(16) __bf16 kstage[4][2][32 * DEP]; // per-wave K double buffer
  const int wave = threadIdx.x >> 5;
  const int lane = threadIdx.x & 31;
  const int wid  = blockIdx.x * 4 + wave;               // 8192 wave tiles
  const int n    = wid >> 7;                            // head-batch 0..63
  const int q0   = (wid & 127) << 4;                    // query tile base
  const int half = lane >> 4, l16 = lane & 15;

  const __bf16* Qb = Qh  + (size_t)n * SS * DEP;
  const __bf16* Kb = Kh  + (size_t)n * SS * DEP;
  const __bf16* Vb = VhT + (size_t)n * DEP * SS;        // (depth, S) rows
  const float*  km = kmask + (size_t)n * SS;

  // async copy geometry: 8 x b128 per 32x64 bf16 tile; lane L moves 16B,
  // row = L>>3 (+4/instr), colByte = (L&7)*16. K row stride = 128B.
  const uint32_t kROW  = (uint32_t)(lane >> 3);
  const uint32_t kCOLB = (uint32_t)((lane & 7) << 4);
  // low 32 bits of a generic LDS pointer == LDS byte address (ISA 10.2)
  const uint32_t kls_base = (uint32_t)(uintptr_t)(&kstage[wave][0][0]);

  // prologue: stage kv tile 0 into buffer 0
  {
    const char* g = (const char*)Kb + (size_t)kROW * 128 + kCOLB;
    const uint32_t l = kls_base + kROW * 128u + kCOLB;
    #pragma unroll
    for (int i = 0; i < 8; ++i) {
      const uint64_t ga = (uint64_t)(uintptr_t)(g + i * 512);
      asm volatile("global_load_async_to_lds_b128 %0, %1, off"
                   :: "v"(l + (uint32_t)(i * 512)), "v"(ga) : "memory");
    }
  }

  // Q A-fragments (K=64 -> two 32-chunks), resident in VGPRs for the whole tile
  v16bf qa[2];
  {
    const __bf16* qrow = Qb + (size_t)(q0 + l16) * DEP;
    #pragma unroll
    for (int kc = 0; kc < 2; ++kc) {
      const int k0 = kc * 32 + half * 16;
      qa[kc] = cat8(*(const v8bf*)(qrow + k0), *(const v8bf*)(qrow + k0 + 8));
    }
  }

  v8f o0 = {}, o1 = {}, o2 = {}, o3 = {};               // O 16x64 accumulators
  float m[8], l[8];
  #pragma unroll
  for (int r = 0; r < 8; ++r) { m[r] = -3.0e38f; l[r] = 0.f; }

  __bf16* pl = &plds[wave][0];
  const float scale = 0.125f;                           // 1/sqrt(64)
  const int kv_end = q0 + 16;                           // causal limit (exclusive)

  int buf = 0;
  for (int kv0 = 0; kv0 < kv_end; kv0 += 32) {
    // ---- V B-fragments issued first: maximal latency window before P@V ----
    const int kvo = kv0 + half * 16;
    v16bf vb0 = *(const v16bf*)(Vb + (size_t)( 0 + l16) * SS + kvo);
    v16bf vb1 = *(const v16bf*)(Vb + (size_t)(16 + l16) * SS + kvo);
    v16bf vb2 = *(const v16bf*)(Vb + (size_t)(32 + l16) * SS + kvo);
    v16bf vb3 = *(const v16bf*)(Vb + (size_t)(48 + l16) * SS + kvo);

    // ---- stream next K tile into the other buffer, wait for current ----
    if (kv0 + 32 < kv_end) {
      const char* g = (const char*)Kb + ((size_t)(kv0 + 32) + kROW) * 128 + kCOLB;
      const uint32_t lb = kls_base + (uint32_t)(buf ^ 1) * (32u * DEP * 2u)
                        + kROW * 128u + kCOLB;
      #pragma unroll
      for (int i = 0; i < 8; ++i) {
        const uint64_t ga = (uint64_t)(uintptr_t)(g + i * 512);
        asm volatile("global_load_async_to_lds_b128 %0, %1, off"
                     :: "v"(lb + (uint32_t)(i * 512)), "v"(ga) : "memory");
      }
      asm volatile("s_wait_asynccnt 0x8" ::: "memory"); // in-order: current tile done
    } else {
      asm volatile("s_wait_asynccnt 0x0" ::: "memory");
    }

    // ---- S = Q @ K^T from LDS-staged K (two 16x16 tiles) ----
    const __bf16* kb = &kstage[wave][buf][0];
    v8f s0 = {}, s1 = {};
    #pragma unroll
    for (int kc = 0; kc < 2; ++kc) {
      const int kofs = kc * 32 + half * 16;
      v16bf b0 = *(const v16bf*)(kb + (size_t)l16 * DEP + kofs);
      s0 = WMMA_BF16(qa[kc], b0, s0);
      v16bf b1 = *(const v16bf*)(kb + (size_t)(16 + l16) * DEP + kofs);
      s1 = WMMA_BF16(qa[kc], b1, s1);
    }

    // ---- scale + causal/key masks + online softmax ----
    const int keyA = kv0 + l16;
    const int keyB = kv0 + 16 + l16;
    const float km0 = km[keyA];
    const float km1 = km[keyB];
    float p0[8], p1[8], al[8];
    #pragma unroll
    for (int r = 0; r < 8; ++r) {
      const int qi = q0 + r + half * 8;                 // row this lane-half owns
      float v0 = s0[r] * scale;
      float v1 = s1[r] * scale;
      if (keyA > qi || km0 == 0.f) v0 = NEGV;
      if (keyB > qi || km1 == 0.f) v1 = NEGV;
      float t = fmaxf(v0, v1);
      #pragma unroll
      for (int off = 1; off < 16; off <<= 1)            // row-max within half-wave
        t = fmaxf(t, __shfl_xor(t, off, 32));
      const float mn = fmaxf(m[r], t);
      al[r] = __expf(m[r] - mn);
      m[r] = mn;
      p0[r] = __expf(v0 - mn);
      p1[r] = __expf(v1 - mn);
    }
    #pragma unroll
    for (int r = 0; r < 8; ++r) {
      float rs = p0[r] + p1[r];
      #pragma unroll
      for (int off = 1; off < 16; off <<= 1)
        rs += __shfl_xor(rs, off, 32);
      l[r] = l[r] * al[r] + rs;
      const int row = r + half * 8;
      pl[row * 40 + l16]      = f2bf(p0[r]);            // P -> LDS (C->A relayout)
      pl[row * 40 + 16 + l16] = f2bf(p1[r]);
      o0[r] *= al[r]; o1[r] *= al[r]; o2[r] *= al[r]; o3[r] *= al[r];
    }

    // ---- P A-fragment from LDS (same-wave DScnt ordering) ----
    v16bf pa;
    {
      const __bf16* prow = pl + l16 * 40 + half * 16;
      pa = cat8(*(const v8bf*)(prow), *(const v8bf*)(prow + 8));
    }

    // ---- O += P @ V ----
    o0 = WMMA_BF16(pa, vb0, o0);
    o1 = WMMA_BF16(pa, vb1, o1);
    o2 = WMMA_BF16(pa, vb2, o2);
    o3 = WMMA_BF16(pa, vb3, o3);

    buf ^= 1;
  }

  // ---- epilogue: divide by l, apply query mask, merge-heads store ----
  const int h = n >> 2, b = n & 3;
  #pragma unroll
  for (int r = 0; r < 8; ++r) {
    const int qi = q0 + r + half * 8;
    const float sc = qmask[(size_t)n * SS + qi] / l[r];
    float* orow = out + ((size_t)b * SS + qi) * DD + h * DEP;
    orow[0 * 16 + l16] = o0[r] * sc;
    orow[1 * 16 + l16] = o1[r] * sc;
    orow[2 * 16 + l16] = o2[r] * sc;
    orow[3 * 16 + l16] = o3[r] * sc;
  }
}

// ---------------------------------------------------------------------------
extern "C" void kernel_launch(void* const* d_in, const int* in_sizes, int n_in,
                              void* d_out, int out_size, void* d_ws, size_t ws_size,
                              hipStream_t stream) {
  const float* q  = (const float*)d_in[0];
  const float* k  = (const float*)d_in[1];
  const float* v  = (const float*)d_in[2];
  const float* Wq = (const float*)d_in[3];
  const float* bq = (const float*)d_in[4];
  const float* Wk = (const float*)d_in[5];
  const float* bk = (const float*)d_in[6];
  const float* Wv = (const float*)d_in[7];
  const float* bv = (const float*)d_in[8];
  float* out = (float*)d_out;

  char* ws = (char*)d_ws;
  const size_t SZ_H = (size_t)NHB * SS * DEP * sizeof(__bf16);   // 16 MB each
  __bf16* Qh   = (__bf16*)(ws);
  __bf16* Kh   = (__bf16*)(ws + SZ_H);
  __bf16* VhT  = (__bf16*)(ws + 2 * SZ_H);
  __bf16* Xbf  = (__bf16*)(ws + 3 * SZ_H);
  __bf16* Wt   = (__bf16*)(ws + 4 * SZ_H);                       // 2 MB, reused
  float*  kmsk = (float*)(ws + 4 * SZ_H + (size_t)DD * DD * sizeof(__bf16));
  float*  qmsk = kmsk + NHB * SS;
  // total workspace: 4*16MB + 2MB + 1MB = 67 MB

  const int nX = BB * SS * DD;                                   // 8388608
  dim3 blk256(256), blk128(128);
  dim3 gcvt((nX / 4 + 255) / 256);
  dim3 gwt(DD / 32, DD / 32);
  dim3 ggemm(8192 / 8);                                          // 8 waves/block
  dim3 gmask((NHB * SS + 255) / 256);
  dim3 gattn((NHB * (SS / 16)) / 4);                             // 4 waves/block

  // Q projection
  cvt_x_kernel<<<gcvt, blk256, 0, stream>>>(q, Xbf, nX);
  cvt_wt_kernel<<<gwt, blk256, 0, stream>>>(Wq, Wt);
  proj_gemm_kernel<<<ggemm, blk256, 0, stream>>>(Xbf, Wt, bq, Qh, 0);
  // K projection
  cvt_x_kernel<<<gcvt, blk256, 0, stream>>>(k, Xbf, nX);
  cvt_wt_kernel<<<gwt, blk256, 0, stream>>>(Wk, Wt);
  proj_gemm_kernel<<<ggemm, blk256, 0, stream>>>(Xbf, Wt, bk, Kh, 0);
  // V projection (stored transposed per head)
  cvt_x_kernel<<<gcvt, blk256, 0, stream>>>(v, Xbf, nX);
  cvt_wt_kernel<<<gwt, blk256, 0, stream>>>(Wv, Wt);
  proj_gemm_kernel<<<ggemm, blk256, 0, stream>>>(Xbf, Wt, bv, VhT, 1);

  mask_kernel<<<gmask, blk256, 0, stream>>>(Qh, Kh, qmsk, kmsk);
  attn_fa_kernel<<<gattn, blk128, 0, stream>>>(Qh, Kh, VhT, kmsk, qmsk, out);
}
